// GraphAttentionRetriever_4544075399870
// MI455X (gfx1250) — matrix operand order repaired and checked by reading, not compiled, not run on hardware
//
#include <hip/hip_runtime.h>
#include <hip/hip_bf16.h>

// ---------------------------------------------------------------------------
// Types for CDNA5 WMMA (wave32, 16x16x32 bf16 -> f32)
// ---------------------------------------------------------------------------
typedef __bf16 bf16;
typedef __attribute__((ext_vector_type(16))) __bf16 v16bf;
typedef __attribute__((ext_vector_type(8)))  __bf16 v8bf;
typedef __attribute__((ext_vector_type(4)))  __bf16 v4bf;
typedef __attribute__((ext_vector_type(8)))  float  v8f;

union Frag { v16bf v; v8bf h[2]; };

__device__ __forceinline__ bf16 f2bf(float f) {
  unsigned u = __builtin_bit_cast(unsigned, f);
  u += 0x7FFFu + ((u >> 16) & 1u);            // round-to-nearest-even
  unsigned short s = (unsigned short)(u >> 16);
  return __builtin_bit_cast(bf16, s);
}

#define BM 128
#define BN 128
#define BK 32
#define LDSK 40   // padded LDS row (bf16 elems) = 80B; multiple of 16B chunks

// ---------------------------------------------------------------------------
// WMMA GEMM, bf16 x bf16 -> f32, async-copy staged, double-buffered LDS.
//   A  : [M,K]  bf16 row-major
//   Bt : [Nc,K] bf16 row-major  (i.e. Y = A @ Bt^T)
//   Yf : optional f32 output [M,ldy];  Ybf: optional bf16 output [M,ldy]
// 256 threads = 8 waves; waves tile 2(M) x 4(N); each wave 4x2 16x16 tiles.
// Staging: GLOBAL_LOAD_ASYNC_TO_LDS_B128 (memory->LDS direct, ASYNCcnt),
// prefetch of tile k+1 overlapped with WMMAs on tile k.
// ---------------------------------------------------------------------------
__global__ __launch_bounds__(256) void gemm_wmma_async(
    const bf16* __restrict__ A, const bf16* __restrict__ Bt,
    float* __restrict__ Yf, bf16* __restrict__ Ybf, const float* __restrict__ bias,
    int M, int K, int Nc, int ldy)
{
  (void)Nc;
  __shared__ __align__(16) bf16 As[2][BM * LDSK];
  __shared__ __align__(16) bf16 Bs[2][BN * LDSK];

  const int tid  = threadIdx.x;
  const int lane = tid & 31;
  const int wv   = tid >> 5;
  const int wy   = wv >> 2;        // 0..1  (M)
  const int wx   = wv & 3;         // 0..3  (N)
  const int m0   = blockIdx.y * BM;
  const int n0   = blockIdx.x * BN;
  const int half = lane >> 4;      // lane half selects K sub-chunk
  const int l15  = lane & 15;

  // LDS byte offsets (flat pointer low 32 bits == LDS address per aperture map)
  const unsigned asb = (unsigned)(size_t)&As[0][0];
  const unsigned bsb = (unsigned)(size_t)&Bs[0][0];

  // stage one BMxBK (and BNxBK) bf16 tile: 512 chunks of 16B each side,
  // 2 chunks per thread per side, issued as async loads to LDS.
  auto stage = [&](int k0, int buf) {
    #pragma unroll
    for (int i = 0; i < 2; ++i) {
      int id = tid + (i << 8);     // 0..511
      int r  = id >> 2;            // row 0..127
      int ch = id & 3;             // 16B chunk within 64B row
      int gr = m0 + r; if (gr >= M) gr = M - 1;   // clamp; results masked at store
      unsigned lA = asb + (unsigned)(buf * (BM * LDSK * 2) + r * (LDSK * 2) + ch * 16);
      unsigned long long gA = (unsigned long long)(size_t)(A + (size_t)gr * K + k0 + ch * 8);
      asm volatile("global_load_async_to_lds_b128 %0, %1, off"
                   :: "v"(lA), "v"(gA) : "memory");
      unsigned lB = bsb + (unsigned)(buf * (BN * LDSK * 2) + r * (LDSK * 2) + ch * 16);
      unsigned long long gB = (unsigned long long)(size_t)(Bt + (size_t)(n0 + r) * K + k0 + ch * 8);
      asm volatile("global_load_async_to_lds_b128 %0, %1, off"
                   :: "v"(lB), "v"(gB) : "memory");
    }
  };

  v8f acc[4][2];
  #pragma unroll
  for (int i = 0; i < 4; ++i)
    #pragma unroll
    for (int j = 0; j < 2; ++j)
      acc[i][j] = v8f{};

  stage(0, 0);
  asm volatile("s_wait_asynccnt 0x0" ::: "memory");
  __syncthreads();

  const int nsteps = K / BK;       // K=256 -> 8
  for (int s = 0; s < nsteps; ++s) {
    const int buf = s & 1;
    if (s + 1 < nsteps) stage((s + 1) * BK, buf ^ 1);   // overlap with WMMAs

    const bf16* AsB = As[buf];
    const bf16* BsB = Bs[buf];

    // fragments per the ISA 16-bit layouts
    Frag af[4];
    #pragma unroll
    for (int mt = 0; mt < 4; ++mt) {
      const bf16* p = &AsB[(wy * 64 + mt * 16 + l15) * LDSK];
      af[mt].h[0] = *(const v8bf*)(p + half * 8);        // K 0..7  / 8..15
      af[mt].h[1] = *(const v8bf*)(p + 16 + half * 8);   // K 16..23 / 24..31
    }
    Frag bfr[2];
    #pragma unroll
    for (int nt = 0; nt < 2; ++nt) {
      const bf16* p = &BsB[(wx * 32 + nt * 16 + l15) * LDSK];
      bfr[nt].h[0] = *(const v8bf*)(p + half * 8);        // K 0..7  / 8..15
      bfr[nt].h[1] = *(const v8bf*)(p + 16 + half * 8);   // K 16..23 / 24..31
    }
    #pragma unroll
    for (int mt = 0; mt < 4; ++mt)
      #pragma unroll
      for (int nt = 0; nt < 2; ++nt)
        acc[mt][nt] = __builtin_amdgcn_wmma_f32_16x16x32_bf16(
            false, af[mt].v, false, bfr[nt].v, (short)0, acc[mt][nt], false, false);

    // all reads of `buf` done; next iteration may overwrite it
    asm volatile("s_wait_asynccnt 0x0" ::: "memory");
    __syncthreads();
  }

  // epilogue: C/D layout (VGPR r: lanes0-15 M=r, lanes16-31 M=r+8)
  #pragma unroll
  for (int mt = 0; mt < 4; ++mt)
    #pragma unroll
    for (int nt = 0; nt < 2; ++nt) {
      int col = n0 + wx * 32 + nt * 16 + l15;
      float bv = bias ? bias[col] : 0.0f;
      #pragma unroll
      for (int r = 0; r < 8; ++r) {
        int row = m0 + wy * 64 + mt * 16 + half * 8 + r;
        if (row < M) {
          float v = acc[mt][nt][r] + bv;
          if (Yf)  Yf[(size_t)row * ldy + col] = v;
          if (Ybf) Ybf[(size_t)row * ldy + col] = f2bf(v);
        }
      }
    }
}

// ---------------------------------------------------------------------------
// One-time fp32 -> bf16 conversion (vectorized 4-wide)
// ---------------------------------------------------------------------------
__global__ void conv_bf16(const float* __restrict__ in, bf16* __restrict__ out, int n4)
{
  int i = blockIdx.x * 256 + threadIdx.x;
  if (i >= n4) return;
  float4 v = *(const float4*)(in + (size_t)i * 4);
  v4bf o = { f2bf(v.x), f2bf(v.y), f2bf(v.z), f2bf(v.w) };
  *(v4bf*)(out + (size_t)i * 4) = o;
}

// transpose + convert: out[z][c*R + r] = in[z][r*Cc + c]   (R=256, Cc=2048)
__global__ __launch_bounds__(256) void transpose_conv(
    const float* __restrict__ in, bf16* __restrict__ out, int R, int Cc)
{
  __shared__ float t[32][33];
  const size_t zoff = (size_t)blockIdx.z * R * Cc;
  int bx = blockIdx.x * 32;   // column
  int by = blockIdx.y * 32;   // row
  int tx = threadIdx.x & 31, ty = threadIdx.x >> 5;  // 32 x 8
  #pragma unroll
  for (int j = 0; j < 32; j += 8)
    t[ty + j][tx] = in[zoff + (size_t)(by + ty + j) * Cc + bx + tx];
  __syncthreads();
  #pragma unroll
  for (int j = 0; j < 32; j += 8)
    out[zoff + (size_t)(bx + ty + j) * R + by + tx] = f2bf(t[tx][ty + j]);
}

// ---------------------------------------------------------------------------
// Per-node attention scores: e_src[n,h] = dot(XT[n,h,:], a_src[h,:]) (len 256)
// ---------------------------------------------------------------------------
__global__ __launch_bounds__(256) void node_scores(
    const float* __restrict__ XT, const float* __restrict__ asrc,
    const float* __restrict__ adst, float* __restrict__ eS, float* __restrict__ eD)
{
  int n = blockIdx.x;
  int h = threadIdx.x >> 5, lane = threadIdx.x & 31;
  const float* xr = XT + (size_t)n * 2048 + h * 256;
  const float* as = asrc + h * 256;
  const float* ad = adst + h * 256;
  float ss = 0.f, sd = 0.f;
  #pragma unroll
  for (int c = lane; c < 256; c += 32) {
    float xv = xr[c];
    ss += xv * as[c];
    sd += xv * ad[c];
  }
  #pragma unroll
  for (int off = 16; off; off >>= 1) {
    ss += __shfl_xor(ss, off, 32);
    sd += __shfl_xor(sd, off, 32);
  }
  if (lane == 0) { eS[n * 8 + h] = ss; eD[n * 8 + h] = sd; }
}

// ---------------------------------------------------------------------------
// Edge softmax pipeline
// ---------------------------------------------------------------------------
__global__ void edge_logits(const int* __restrict__ src, const int* __restrict__ dst,
                            const float* __restrict__ eS, const float* __restrict__ eD,
                            float* __restrict__ eV, unsigned* __restrict__ mO, int E)
{
  int idx = blockIdx.x * 256 + threadIdx.x;
  if (idx >= E * 8) return;
  int e = idx >> 3, h = idx & 7;
  int s = src[e], d = dst[e];
  float v = eS[s * 8 + h] + eD[d * 8 + h];
  v = v > 0.f ? v : 0.2f * v;
  eV[idx] = v;
  unsigned u = __builtin_bit_cast(unsigned, v);
  unsigned o = (u & 0x80000000u) ? ~u : (u | 0x80000000u);
  atomicMax(&mO[d * 8 + h], o);
}

__global__ void edge_exp(const int* __restrict__ dst, float* __restrict__ eV,
                         const unsigned* __restrict__ mO, float* __restrict__ den, int E)
{
  int idx = blockIdx.x * 256 + threadIdx.x;
  if (idx >= E * 8) return;
  int e = idx >> 3, h = idx & 7;
  int d = dst[e];
  unsigned o = mO[d * 8 + h];
  float m = 0.f;
  if (o) {
    unsigned u = (o & 0x80000000u) ? (o & 0x7FFFFFFFu) : ~o;
    m = __builtin_bit_cast(float, u);
  }
  float ee = __expf(eV[idx] - m);
  eV[idx] = ee;
  atomicAdd(&den[d * 8 + h], ee);
}

__global__ __launch_bounds__(256) void edge_scatter(
    const int* __restrict__ src, const int* __restrict__ dst,
    const float* __restrict__ eV, const float* __restrict__ den,
    const float* __restrict__ XT, float* __restrict__ AGG, int E)
{
  int e = blockIdx.x;
  __shared__ float sa[8];
  int d = dst[e];
  if (threadIdx.x < 8)
    sa[threadIdx.x] = eV[(size_t)e * 8 + threadIdx.x] / (den[d * 8 + threadIdx.x] + 1e-16f);
  __syncthreads();
  int s = src[e];
  int c = threadIdx.x;
  const float* xr = XT + (size_t)s * 2048;
  float* ar = AGG + (size_t)d * 2048;
  #pragma unroll
  for (int h = 0; h < 8; ++h)
    atomicAdd(ar + h * 256 + c, sa[h] * xr[h * 256 + c]);
}

__global__ __launch_bounds__(256) void mean_bias_acc(
    const float* __restrict__ AGG, const float* __restrict__ gb, float* __restrict__ acc)
{
  int n = blockIdx.x, c = threadIdx.x;
  const float* ar = AGG + (size_t)n * 2048 + c;
  float s = 0.f;
  #pragma unroll
  for (int h = 0; h < 8; ++h) s += ar[h * 256];
  acc[(size_t)n * 256 + c] += s * 0.125f + gb[c];
}

// relu straight into bf16 (next consumer is a WMMA GEMM)
__global__ void relu_bf16(const float* __restrict__ in, bf16* __restrict__ out, int n)
{
  int i = blockIdx.x * 256 + threadIdx.x;
  if (i < n) out[i] = f2bf(fmaxf(in[i], 0.f));
}

// ---------------------------------------------------------------------------
// Launcher
// ---------------------------------------------------------------------------
extern "C" void kernel_launch(void* const* d_in, const int* in_sizes, int n_in,
                              void* d_out, int out_size, void* d_ws, size_t ws_size,
                              hipStream_t stream)
{
  (void)n_in; (void)out_size; (void)ws_size;
  const float* x       = (const float*)d_in[0];
  const int*   ei0     = (const int*)d_in[1];
  const int*   ei1     = (const int*)d_in[2];
  const float* gat_W   = (const float*)d_in[3];
  const float* att_src = (const float*)d_in[4];
  const float* att_dst = (const float*)d_in[5];
  const float* gat_b   = (const float*)d_in[6];
  const float* in_w    = (const float*)d_in[7];
  const float* in_b    = (const float*)d_in[8];
  const float* out_w   = (const float*)d_in[9];
  const float* out_b   = (const float*)d_in[10];
  const float* proj_w  = (const float*)d_in[11];
  const float* proj_b  = (const float*)d_in[12];

  const int C = 256, H = 8, L = 3, T = 2;
  const int N  = in_sizes[0] / C;
  const int E  = in_sizes[1] / 2;
  const int HC = H * C;   // 2048

  char* w = (char*)d_ws;
  auto alloc = [&](size_t bytes) -> void* {
    void* p = (void*)w;
    w += (bytes + 255) & ~(size_t)255;
    return p;
  };
  float*    hAcc  = (float*)alloc((size_t)N * C * 4);
  float*    XT    = (float*)alloc((size_t)N * HC * 4);
  float*    AGG   = (float*)alloc((size_t)N * HC * 4);
  float*    eS    = (float*)alloc((size_t)N * H * 4);
  float*    eD    = (float*)alloc((size_t)N * H * 4);
  unsigned* mO    = (unsigned*)alloc((size_t)N * H * 4);
  float*    den   = (float*)alloc((size_t)N * H * 4);
  float*    eV    = (float*)alloc((size_t)E * H * 4);
  bf16*     hbf   = (bf16*)alloc((size_t)N * C * 2);        // x / relu(h), bf16
  bf16*     WtAll = (bf16*)alloc((size_t)L * T * HC * C * 2); // gat_W^T, bf16 [lt][2048][256]
  bf16*     wv_bf = (bf16*)alloc((size_t)C * C * 2);
  bf16*     wo_bf = (bf16*)alloc((size_t)C * C * 2);
  bf16*     wp_bf = (bf16*)alloc((size_t)C * C * 2);
  bf16*     Vbf   = (bf16*)alloc((size_t)N * C * 2);
  bf16*     O1bf  = (bf16*)alloc((size_t)N * C * 2);

  // ---- one-time conversions (bf16 operands for all WMMA GEMMs) ----
  conv_bf16<<<((N * C / 4) + 255) / 256, 256, 0, stream>>>(x, hbf, N * C / 4);
  {
    dim3 gt(HC / 32, C / 32, L * T);
    transpose_conv<<<gt, 256, 0, stream>>>(gat_W, WtAll, C, HC);
  }
  conv_bf16<<<((C * C / 4) + 255) / 256, 256, 0, stream>>>(in_w + (size_t)2 * C * C, wv_bf, C * C / 4);
  conv_bf16<<<((C * C / 4) + 255) / 256, 256, 0, stream>>>(out_w, wo_bf, C * C / 4);
  conv_bf16<<<((C * C / 4) + 255) / 256, 256, 0, stream>>>(proj_w, wp_bf, C * C / 4);

  for (int l = 0; l < L; ++l) {
    hipMemsetAsync(hAcc, 0, (size_t)N * C * 4, stream);
    for (int t = 0; t < T; ++t) {
      const int lt = l * T + t;
      // XT = h @ W[l,t]   [N,256] x [256,2048]  (B pre-transposed to [2048][256])
      dim3 gg(HC / BN, (N + BM - 1) / BM);
      gemm_wmma_async<<<gg, 256, 0, stream>>>(
          hbf, WtAll + (size_t)lt * HC * C, XT, nullptr, nullptr, N, C, HC, HC);

      hipMemsetAsync(AGG, 0, (size_t)N * HC * 4, stream);
      hipMemsetAsync(mO,  0, (size_t)N * H * 4, stream);
      hipMemsetAsync(den, 0, (size_t)N * H * 4, stream);

      node_scores<<<N, 256, 0, stream>>>(
          XT, att_src + (size_t)lt * H * C, att_dst + (size_t)lt * H * C, eS, eD);

      const int* src = (t == 0 ? ei0 : ei1);
      const int* dst = src + E;
      int eth = E * H;
      edge_logits<<<(eth + 255) / 256, 256, 0, stream>>>(src, dst, eS, eD, eV, mO, E);
      edge_exp   <<<(eth + 255) / 256, 256, 0, stream>>>(dst, eV, mO, den, E);
      edge_scatter<<<E, 256, 0, stream>>>(src, dst, eV, den, XT, AGG, E);
      mean_bias_acc<<<N, 256, 0, stream>>>(AGG, gat_b + (size_t)lt * C, hAcc);
    }
    relu_bf16<<<((N * C) + 255) / 256, 256, 0, stream>>>(hAcc, hbf, N * C);
  }

  // Final MHA with seq_len=1: softmax == 1, only the V path matters.
  // v = h @ in_w[2C:3C].T + in_b[2C:] ; out = (v @ out_w.T + out_b) @ proj_w.T + proj_b
  dim3 gl(C / BN, (N + BM - 1) / BM);
  gemm_wmma_async<<<gl, 256, 0, stream>>>(
      hbf, wv_bf, nullptr, Vbf, in_b + 2 * C, N, C, C, C);
  gemm_wmma_async<<<gl, 256, 0, stream>>>(
      Vbf, wo_bf, nullptr, O1bf, out_b, N, C, C, C);
  gemm_wmma_async<<<gl, 256, 0, stream>>>(
      O1bf, wp_bf, (float*)d_out, nullptr, proj_b, N, C, C, C);
}
